// RobustPrompt_I_61924838473938
// MI455X (gfx1250) — matrix operand order repaired and checked by reading, not compiled, not run on hardware
//
#include <hip/hip_runtime.h>
#include <hip/hip_bf16.h>

// ---------------------------------------------------------------------------
// RobustPrompt graph-prompt pipeline for MI455X (gfx1250, wave32, WMMA).
//
// Roofline: HBM traffic ~110MB (x in/out + edges + masks) ≈ 5us at 23.3TB/s.
// The 2.6GB of per-edge row gathers are L2-resident (26MB f16 table vs 192MB
// L2). Edge cosines: 16 at a time as the diagonal of v_wmma_f32_16x16x32_f16
// over K=128; gathers go straight to LDS via the gfx1250 async-DMA path
// (GLOBAL_LOAD_ASYNC_TO_LDS_B128, ASYNCcnt) when the builtin is available.
// All LDS traffic is wave-private -> no block barriers needed (per-wave LDS
// ordering is architectural).
// ---------------------------------------------------------------------------

typedef __attribute__((ext_vector_type(16))) _Float16 v16h;
typedef __attribute__((ext_vector_type(8)))  _Float16 v8h;
typedef __attribute__((ext_vector_type(4)))  _Float16 v4h;
typedef __attribute__((ext_vector_type(8)))  float    v8f;
typedef __attribute__((ext_vector_type(4)))  int      v4i;

#define DIMS     128
#define SIM_THR  0.2f
#define DEG_THR  3.0f
#define PT_THR   0.1f
#define EPSV     1e-8f
#define WPB      4      // waves per block in the edge kernel (36KB LDS/block)

__device__ __forceinline__ float waveReduceSum(float v) {
#pragma unroll
    for (int off = 16; off > 0; off >>= 1)
        v += __shfl_xor(v, off, 32);
    return v;
}

// --- kernel 1: x_norm = x / ||x||, stored fp16 (one wave per row) ----------
__global__ void rp_normalize_f16(const float* __restrict__ x,
                                 _Float16* __restrict__ xf, int n) {
    int row  = blockIdx.x * 8 + (threadIdx.x >> 5);
    int lane = threadIdx.x & 31;
    if (row >= n) return;
    float4 xv = ((const float4*)(x + (size_t)row * DIMS))[lane];
    float ss = xv.x * xv.x + xv.y * xv.y + xv.z * xv.z + xv.w * xv.w;
    ss = waveReduceSum(ss);
    float inv = rsqrtf(ss);                       // reference: no eps in pass 1
    v4h h = { (_Float16)(xv.x * inv), (_Float16)(xv.y * inv),
              (_Float16)(xv.z * inv), (_Float16)(xv.w * inv) };
    ((v4h*)(xf + (size_t)row * DIMS))[lane] = h;
}

// --- gather one 256B row into this lane's LDS tile row ---------------------
#if __has_builtin(__builtin_amdgcn_global_load_async_to_lds_b128)
typedef __attribute__((address_space(1))) v4i g_v4i;   // AS1: global (__device__)
typedef __attribute__((address_space(3))) v4i l_v4i;   // AS3: LDS (__shared__)
#define RP_HAS_ASYNC_LDS 1
#else
#define RP_HAS_ASYNC_LDS 0
#endif

__device__ __forceinline__ void rp_gather_row(const _Float16* __restrict__ src,
                                              _Float16* __restrict__ dst) {
#if RP_HAS_ASYNC_LDS
    const char* sp = (const char*)src;
    char*       dp = (char*)dst;
#pragma unroll
    for (int i = 0; i < 16; ++i) {                 // 16 x 16B async DMA to LDS
        __builtin_amdgcn_global_load_async_to_lds_b128(
            (g_v4i*)(const void*)sp, (l_v4i*)(void*)dp, 0, 0);
        sp += 16; dp += 16;
    }
#else
    const uint4* sp = (const uint4*)src;
    uint4*       dp = (uint4*)dst;
#pragma unroll
    for (int i = 0; i < 16; ++i) dp[i] = sp[i];    // VGPR-staged fallback
#endif
}

__device__ __forceinline__ void rp_gather_wait() {
#if RP_HAS_ASYNC_LDS
#if __has_builtin(__builtin_amdgcn_s_wait_asynccnt)
    __builtin_amdgcn_s_wait_asynccnt(0);
#else
    asm volatile("s_wait_asynccnt 0x0" ::: "memory");
#endif
#endif
    // per-wave LDS ordering is architectural; just stop IR-level motion
    asm volatile("" ::: "memory");
}

// --- kernel 2/4: 16 edge cosines per wave via WMMA diagonal ----------------
// MODE 0: atomic scatter cos -> c[col], 1 -> deg[col]
// MODE 1: keep[e] = (cos >= PT_THR)
template <int MODE>
__global__ void rp_edge_cos_wmma(const _Float16* __restrict__ xf,
                                 const int* __restrict__ rows,
                                 const int* __restrict__ cols,
                                 float* __restrict__ cacc,
                                 float* __restrict__ dacc,
                                 float* __restrict__ keep,
                                 int E) {
    __shared__ __align__(16) _Float16 tile[WPB][32][DIMS];   // 32 gathered rows / wave
    __shared__ float dmat[WPB][16][16];                      // WMMA D bounce (wave-private)

    const int slot = threadIdx.x >> 5;
    const int lane = threadIdx.x & 31;
    const int r    = lane & 15;

    int ebase = (blockIdx.x * WPB + slot) * 16;
    int myE   = ebase + r;                 // clamp tail so EXEC stays all-1s
    if (myE > E - 1) myE = E - 1;

    // lanes 0-15 gather the A rows (src endpoints), 16-31 the B rows (dst).
    const int src = (lane < 16) ? rows[myE] : cols[myE];
    rp_gather_row(xf + (size_t)src * DIMS, &tile[slot][lane][0]);
    rp_gather_wait();

    // Build the wave32 16-bit A (16x32) and B (32x16) VGPR stripings from LDS.
    const int aLo = (lane < 16) ? 0 : 8;    // A: K {0-7,16-23} vs {8-15,24-31}
    const int bLo = (lane < 16) ? 0 : 8;    // B: K {0-7,16-23} vs {8-15,24-31}
    v8f acc = {};
#pragma unroll
    for (int k0 = 0; k0 < DIMS; k0 += 32) {
        v8h a0 = *(const v8h*)&tile[slot][r][k0 + aLo];
        v8h a1 = *(const v8h*)&tile[slot][r][k0 + aLo + 16];
        v16h A = __builtin_shufflevector(a0, a1, 0, 1, 2, 3, 4, 5, 6, 7,
                                                 8, 9, 10, 11, 12, 13, 14, 15);
        v8h b0 = *(const v8h*)&tile[slot][16 + r][k0 + bLo];
        v8h b1 = *(const v8h*)&tile[slot][16 + r][k0 + bLo + 16];
        v16h B = __builtin_shufflevector(b0, b1, 0, 1, 2, 3, 4, 5, 6, 7,
                                                 8, 9, 10, 11, 12, 13, 14, 15);
        acc = __builtin_amdgcn_wmma_f32_16x16x32_f16(
                  false, A, false, B, (short)0, acc, false, false);
    }

    // D layout: lanes 0-15 hold M=vgpr,N=lane; lanes 16-31 hold M=8+vgpr,N=lane-16.
    const int mb = (lane < 16) ? 0 : 8;
    dmat[slot][mb + 0][r] = acc[0];
    dmat[slot][mb + 1][r] = acc[1];
    dmat[slot][mb + 2][r] = acc[2];
    dmat[slot][mb + 3][r] = acc[3];
    dmat[slot][mb + 4][r] = acc[4];
    dmat[slot][mb + 5][r] = acc[5];
    dmat[slot][mb + 6][r] = acc[6];
    dmat[slot][mb + 7][r] = acc[7];
    asm volatile("" ::: "memory");           // wave-private LDS: hw keeps order

    if (lane >= 16) {                        // lane 16+m owns edge m (has col id)
        int m = lane - 16;
        int eg = ebase + m;
        if (eg < E) {
            float cosv = dmat[slot][m][m];
            if (MODE == 0) {
                atomicAdd(&cacc[src], cosv);
                atomicAdd(&dacc[src], 1.0f);
            } else {
                keep[eg] = (cosv >= PT_THR) ? 1.0f : 0.0f;
            }
        }
    }
}

// --- kernel 3: masks, prompt blend, x_new, re-normalized fp16 --------------
__global__ void rp_node_prompt(const float* __restrict__ x,
                               const float* __restrict__ cacc,
                               const float* __restrict__ dacc,
                               const float* __restrict__ p_sim,
                               const float* __restrict__ p_deg,
                               const float* __restrict__ p_other,
                               float* __restrict__ xout,
                               _Float16* __restrict__ xf, int n) {
    int row  = blockIdx.x * 8 + (threadIdx.x >> 5);
    int lane = threadIdx.x & 31;
    if (row >= n) return;

    float cv = cacc[row];
    float dv = dacc[row];
    bool ms = (dv > 0.0f) && ((cv / fmaxf(dv, 1.0f)) <= SIM_THR); // deg==0 -> inf -> false
    bool md = (dv <= DEG_THR);
    bool mo = !(ms || md);
    float fs = ms ? 1.0f : 0.0f, fd = md ? 1.0f : 0.0f, fo = mo ? 1.0f : 0.0f;
    float invp = 1.0f / (fs + fd + fo);                           // plen >= 1 always

    float4 xv = ((const float4*)(x + (size_t)row * DIMS))[lane];
    float4 ps = ((const float4*)p_sim)[lane];
    float4 pd = ((const float4*)p_deg)[lane];
    float4 po = ((const float4*)p_other)[lane];
    float4 xn;
    xn.x = xv.x + (fs * ps.x + fd * pd.x + fo * po.x) * invp;
    xn.y = xv.y + (fs * ps.y + fd * pd.y + fo * po.y) * invp;
    xn.z = xv.z + (fs * ps.z + fd * pd.z + fo * po.z) * invp;
    xn.w = xv.w + (fs * ps.w + fd * pd.w + fo * po.w) * invp;
    ((float4*)(xout + (size_t)row * DIMS))[lane] = xn;

    float ss = xn.x * xn.x + xn.y * xn.y + xn.z * xn.z + xn.w * xn.w;
    ss = waveReduceSum(ss);
    float rn = 1.0f / fmaxf(sqrtf(ss), EPSV);                     // cosine eps clamp
    v4h h = { (_Float16)(xn.x * rn), (_Float16)(xn.y * rn),
              (_Float16)(xn.z * rn), (_Float16)(xn.w * rn) };
    ((v4h*)(xf + (size_t)row * DIMS))[lane] = h;
}

extern "C" void kernel_launch(void* const* d_in, const int* in_sizes, int n_in,
                              void* d_out, int out_size, void* d_ws, size_t ws_size,
                              hipStream_t stream) {
    const float* x       = (const float*)d_in[0];
    const int*   ei      = (const int*)d_in[1];
    const float* p_sim   = (const float*)d_in[2];
    const float* p_deg   = (const float*)d_in[3];
    const float* p_other = (const float*)d_in[4];

    const int n = in_sizes[0] / DIMS;
    const int E = in_sizes[1] / 2;
    const int* rows = ei;
    const int* cols = ei + E;

    // workspace: fp16 feature table (reused for x_norm then x_new_hat) + c/deg
    char* ws = (char*)d_ws;
    _Float16* xf = (_Float16*)ws;
    size_t off = (((size_t)n * DIMS * sizeof(_Float16)) + 255) & ~(size_t)255;
    float* cbuf = (float*)(ws + off);
    float* dbuf = cbuf + n;

    float* xout = (float*)d_out;
    float* keep = xout + (size_t)n * DIMS;

    (void)hipMemsetAsync(cbuf, 0, 2 * (size_t)n * sizeof(float), stream);

    const int nodeBlocks = (n + 7) / 8;                      // 1 wave/row, 8 waves/block
    const int edgeBlocks = (E + 16 * WPB - 1) / (16 * WPB);  // 16 edges/wave

    rp_normalize_f16<<<nodeBlocks, 256, 0, stream>>>(x, xf, n);
    rp_edge_cos_wmma<0><<<edgeBlocks, 32 * WPB, 0, stream>>>(xf, rows, cols,
                                                             cbuf, dbuf, keep, E);
    rp_node_prompt<<<nodeBlocks, 256, 0, stream>>>(x, cbuf, dbuf,
                                                   p_sim, p_deg, p_other,
                                                   xout, xf, n);
    rp_edge_cos_wmma<1><<<edgeBlocks, 32 * WPB, 0, stream>>>(xf, rows, cols,
                                                             cbuf, dbuf, keep, E);
}
